// Qwen2MoeForCausalLM_53042846105772
// MI455X (gfx1250) — matrix operand order repaired and checked by reading, compiled, not run on hardware
//
#include <hip/hip_runtime.h>
#include <math.h>

#define NTOK   2048
#define DMODEL 2048
#define NH     16
#define DHEAD  128
#define DQKV   2048   /* NH*DHEAD */
#define NEXP   8
#define DFF    1408
#define DFFS   5632
#define NLAYER 2
#define EPSV   1e-6f

typedef __attribute__((ext_vector_type(16))) __bf16 bf16x16;
typedef __attribute__((ext_vector_type(8)))  __bf16 bf16x8;
typedef __attribute__((ext_vector_type(8)))  float  f32x8;

__device__ __forceinline__ unsigned short f2bf(float f) {
  unsigned int u = __float_as_uint(f);
  unsigned int r = (u + 0x7fffu + ((u >> 16) & 1u)) >> 16;   // RNE
  return (unsigned short)r;
}
__device__ __forceinline__ float bf2f(unsigned short b) {
  return __uint_as_float(((unsigned int)b) << 16);
}

// ---------------------------------------------------------------------------
// C(MxN, f32) = A(MxK, bf16 row-major) * Bt(NxK, bf16 "B transposed")
// One wave per 32x32 C macro-tile (2x2 register blocking): per K-step of 32,
// load 2 A-frags + 2 B-frags (8x b128) and issue 4 v_wmma_f32_16x16x32_bf16
// with independent accumulators (2 loads per WMMA, 4-deep chain interleave).
// Fragment layouts follow CDNA5 ISA 7.12.2:
//   A 16x32 : lanes 0-15 = rows, elems 0..7 -> K=kg*8+0..7, elems 8..15 -> K=16+kg*8+..
//   B 32x16 : lanes 0-15 = cols (N), elems 0..7 -> K=kg*8+0..7, elems 8..15 -> K=16+kg*8+.. (as A)
//   C 16x16 : vgpr r -> M = kg*8 + r, N = lane&15
// ---------------------------------------------------------------------------
__global__ void __launch_bounds__(256)
gemm_bf16_wmma(const unsigned short* __restrict__ A, int lda,
               const unsigned short* __restrict__ Bt, int ldb,
               float* __restrict__ C, int ldc,
               int M, int N, int Kd)
{
  const int wid    = threadIdx.x >> 5;
  const int lane   = threadIdx.x & 31;
  const int tilesN = N >> 5;                     // 32-wide macro tiles
  const int tilesM = M >> 5;
  const int tile   = blockIdx.x * 8 + wid;
  if (tile >= tilesN * tilesM) return;           // wave-uniform: EXEC stays all-ones
  const int tm  = tile / tilesN;
  const int tn  = tile % tilesN;
  const int l16 = lane & 15;
  const int kg  = lane >> 4;

  const __bf16* Ab  = reinterpret_cast<const __bf16*>(A);
  const __bf16* Btb = reinterpret_cast<const __bf16*>(Bt);
  const __bf16* a0 = Ab  + (size_t)(tm * 32 + l16) * lda + kg * 8;
  const __bf16* a1 = a0 + (size_t)16 * lda;
  const __bf16* b0 = Btb + (size_t)(tn * 32 + l16) * ldb + kg * 8;
  const __bf16* b1 = b0 + (size_t)16 * ldb;

  f32x8 acc00 = {}, acc01 = {}, acc10 = {}, acc11 = {};
  for (int k = 0; k < Kd; k += 32) {
    __builtin_prefetch(a0 + k + 512, 0, 0);
    __builtin_prefetch(b0 + k + 512, 0, 0);
    bf16x8 a0lo = *reinterpret_cast<const bf16x8*>(a0 + k);
    bf16x8 a0hi = *reinterpret_cast<const bf16x8*>(a0 + k + 16);
    bf16x8 a1lo = *reinterpret_cast<const bf16x8*>(a1 + k);
    bf16x8 a1hi = *reinterpret_cast<const bf16x8*>(a1 + k + 16);
    bf16x8 b0lo = *reinterpret_cast<const bf16x8*>(b0 + k);
    bf16x8 b0hi = *reinterpret_cast<const bf16x8*>(b0 + k + 16);
    bf16x8 b1lo = *reinterpret_cast<const bf16x8*>(b1 + k);
    bf16x8 b1hi = *reinterpret_cast<const bf16x8*>(b1 + k + 16);
    bf16x16 fa0, fa1, fb0, fb1;
#pragma unroll
    for (int i = 0; i < 8; ++i) {
      fa0[i] = a0lo[i]; fa0[i + 8] = a0hi[i];
      fa1[i] = a1lo[i]; fa1[i + 8] = a1hi[i];
      fb0[i] = b0lo[i]; fb0[i + 8] = b0hi[i];
      fb1[i] = b1lo[i]; fb1[i + 8] = b1hi[i];
    }
    acc00 = __builtin_amdgcn_wmma_f32_16x16x32_bf16(false, fa0, false, fb0, (short)0, acc00, false, false);
    acc01 = __builtin_amdgcn_wmma_f32_16x16x32_bf16(false, fa0, false, fb1, (short)0, acc01, false, false);
    acc10 = __builtin_amdgcn_wmma_f32_16x16x32_bf16(false, fa1, false, fb0, (short)0, acc10, false, false);
    acc11 = __builtin_amdgcn_wmma_f32_16x16x32_bf16(false, fa1, false, fb1, (short)0, acc11, false, false);
  }
  float* c00 = C + (size_t)(tm * 32 + kg * 8) * ldc + tn * 32 + l16;
  float* c01 = c00 + 16;
  float* c10 = c00 + (size_t)16 * ldc;
  float* c11 = c10 + 16;
#pragma unroll
  for (int r = 0; r < 8; ++r) {
    c00[(size_t)r * ldc] = acc00[r];
    c01[(size_t)r * ldc] = acc01[r];
    c10[(size_t)r * ldc] = acc10[r];
    c11[(size_t)r * ldc] = acc11[r];
  }
}

// Transpose + f32->bf16: dst[c*rows + r] = bf16(src[r*cols + c])  (LDS tiled)
__global__ void transpose_cvt(const float* __restrict__ src,
                              unsigned short* __restrict__ dst,
                              int rows, int cols)
{
  __shared__ float tile[32][33];
  const int cb = blockIdx.x * 32, rb = blockIdx.y * 32;
  const int tx = threadIdx.x, ty = threadIdx.y;
  for (int j = ty; j < 32; j += 8) {
    int r = rb + j, c = cb + tx;
    if (r < rows && c < cols) tile[j][tx] = src[(size_t)r * cols + c];
  }
  __syncthreads();
  for (int j = ty; j < 32; j += 8) {
    int c = cb + j, r = rb + tx;
    if (c < cols && r < rows) dst[(size_t)c * rows + r] = f2bf(tile[tx][j]);
  }
}

__global__ void cvt_f32_bf16(const float* __restrict__ in,
                             unsigned short* __restrict__ out, size_t n)
{
  size_t i = (size_t)blockIdx.x * blockDim.x + threadIdx.x;
  size_t stride = (size_t)gridDim.x * blockDim.x;
  for (; i < n; i += stride) out[i] = f2bf(in[i]);
}

__global__ void embed_gather(const int* __restrict__ ids,
                             const float* __restrict__ emb,
                             float* __restrict__ resid)
{
  const int t = blockIdx.x;
  const float* row = emb + (size_t)ids[t] * DMODEL;
  for (int d = threadIdx.x; d < DMODEL; d += blockDim.x)
    resid[(size_t)t * DMODEL + d] = row[d];
}

__global__ void __launch_bounds__(256)
rmsnorm(const float* __restrict__ in, const float* __restrict__ w,
        unsigned short* __restrict__ obf, float* __restrict__ of32)
{
  __shared__ float red[256];
  const int t = blockIdx.x, tid = threadIdx.x;
  const float* row = in + (size_t)t * DMODEL;
  float ss = 0.f;
  for (int d = tid; d < DMODEL; d += 256) { float v = row[d]; ss += v * v; }
  red[tid] = ss; __syncthreads();
  for (int s = 128; s > 0; s >>= 1) { if (tid < s) red[tid] += red[tid + s]; __syncthreads(); }
  const float inv = rsqrtf(red[0] / (float)DMODEL + EPSV);
  for (int d = tid; d < DMODEL; d += 256) {
    float v = row[d] * inv * w[d];
    if (obf)  obf[(size_t)t * DMODEL + d]  = f2bf(v);
    if (of32) of32[(size_t)t * DMODEL + d] = v;
  }
}

__global__ void rope_cvt(const float* __restrict__ qf, const float* __restrict__ kf,
                         const int* __restrict__ pos,
                         unsigned short* __restrict__ qbf, unsigned short* __restrict__ kbf)
{
  const int th = blockIdx.x;               // t*NH + h
  const int t = th >> 4;
  const int d = threadIdx.x;               // blockDim = 64 = DHEAD/2
  const float p = (float)pos[t];
  const float freq = __expf(-((float)d / 64.f) * 13.815510557964274f);  // ln(1e6)
  const float ang = p * freq;
  const float c = cosf(ang), s = sinf(ang);
  const size_t base = (size_t)th * DHEAD;
  float q1 = qf[base + d], q2 = qf[base + 64 + d];
  qbf[base + d]      = f2bf(q1 * c - q2 * s);
  qbf[base + 64 + d] = f2bf(q1 * s + q2 * c);
  float k1 = kf[base + d], k2 = kf[base + 64 + d];
  kbf[base + d]      = f2bf(k1 * c - k2 * s);
  kbf[base + 64 + d] = f2bf(k1 * s + k2 * c);
}

__global__ void __launch_bounds__(256)
softmax_causal(const float* __restrict__ scores, unsigned short* __restrict__ pbf)
{
  __shared__ float red[256];
  const int t = blockIdx.x, tid = threadIdx.x;
  const float scale = 0.08838834764831845f;     // 1/sqrt(128)
  const float* row = scores + (size_t)t * NTOK;
  float mx = -1e30f;
  for (int s = tid; s <= t; s += 256) mx = fmaxf(mx, row[s] * scale);
  red[tid] = mx; __syncthreads();
  for (int s = 128; s > 0; s >>= 1) { if (tid < s) red[tid] = fmaxf(red[tid], red[tid + s]); __syncthreads(); }
  const float m = red[0]; __syncthreads();
  float sum = 0.f;
  for (int s = tid; s <= t; s += 256) sum += __expf(row[s] * scale - m);
  red[tid] = sum; __syncthreads();
  for (int s = 128; s > 0; s >>= 1) { if (tid < s) red[tid] += red[tid + s]; __syncthreads(); }
  const float z = 1.f / red[0];
  for (int s = tid; s < NTOK; s += 256) {
    float p = (s <= t) ? __expf(row[s] * scale - m) * z : 0.f;
    pbf[(size_t)t * NTOK + s] = f2bf(p);
  }
}

__global__ void __launch_bounds__(256)
router_kernel(const unsigned short* __restrict__ xbf, const float* __restrict__ rw,
              const float* __restrict__ sgw, float* __restrict__ combine,
              float* __restrict__ sgate)
{
  __shared__ float red[256];
  __shared__ float outv[NEXP + 1];
  const int t = blockIdx.x, tid = threadIdx.x;
  float part[NEXP + 1];
#pragma unroll
  for (int e = 0; e <= NEXP; ++e) part[e] = 0.f;
  for (int d = tid; d < DMODEL; d += 256) {
    float x = bf2f(xbf[(size_t)t * DMODEL + d]);
#pragma unroll
    for (int e = 0; e < NEXP; ++e) part[e] += x * rw[(size_t)d * NEXP + e];
    part[NEXP] += x * sgw[d];
  }
  for (int ch = 0; ch <= NEXP; ++ch) {
    red[tid] = part[ch]; __syncthreads();
    for (int s = 128; s > 0; s >>= 1) { if (tid < s) red[tid] += red[tid + s]; __syncthreads(); }
    if (tid == 0) outv[ch] = red[0];
    __syncthreads();
  }
  if (tid == 0) {
    float mx = outv[0];
    for (int e = 1; e < NEXP; ++e) mx = fmaxf(mx, outv[e]);
    float p[NEXP]; float sum = 0.f;
    for (int e = 0; e < NEXP; ++e) { p[e] = __expf(outv[e] - mx); sum += p[e]; }
    for (int e = 0; e < NEXP; ++e) p[e] /= sum;
    int i0 = 0;
    for (int e = 1; e < NEXP; ++e) if (p[e] > p[i0]) i0 = e;
    int i1 = (i0 == 0) ? 1 : 0;
    for (int e = 0; e < NEXP; ++e) if (e != i0 && p[e] > p[i1]) i1 = e;
    const float norm = p[i0] + p[i1];
    for (int e = 0; e < NEXP; ++e) {
      float v = (e == i0) ? p[i0] / norm : (e == i1) ? p[i1] / norm : 0.f;
      combine[(size_t)t * NEXP + e] = v;
    }
    sgate[t] = 1.f / (1.f + __expf(-outv[NEXP]));
  }
}

__global__ void silu_mul(const float* __restrict__ g, const float* __restrict__ u,
                         unsigned short* __restrict__ out, size_t n)
{
  size_t i = (size_t)blockIdx.x * blockDim.x + threadIdx.x;
  size_t stride = (size_t)gridDim.x * blockDim.x;
  for (; i < n; i += stride) {
    float gv = g[i];
    out[i] = f2bf(gv / (1.f + __expf(-gv)) * u[i]);
  }
}

__global__ void add_inplace(float* __restrict__ dst, const float* __restrict__ src, size_t n)
{
  size_t i = (size_t)blockIdx.x * blockDim.x + threadIdx.x;
  size_t stride = (size_t)gridDim.x * blockDim.x;
  for (; i < n; i += stride) dst[i] += src[i];
}

__global__ void moe_accum(float* __restrict__ acc, const float* __restrict__ y,
                          const float* __restrict__ combine, int e, int first)
{
  const size_t n = (size_t)NTOK * DMODEL;
  size_t i = (size_t)blockIdx.x * blockDim.x + threadIdx.x;
  size_t stride = (size_t)gridDim.x * blockDim.x;
  for (; i < n; i += stride) {
    int t = (int)(i / DMODEL);
    float v = combine[(size_t)t * NEXP + e] * y[i];
    acc[i] = first ? v : acc[i] + v;
  }
}

__global__ void moe_final(float* __restrict__ resid, const float* __restrict__ acc,
                          const float* __restrict__ sh, const float* __restrict__ sgate)
{
  const float invs = 0.7071067811865476f;   // 1/sqrt(TOPK)
  const size_t n = (size_t)NTOK * DMODEL;
  size_t i = (size_t)blockIdx.x * blockDim.x + threadIdx.x;
  size_t stride = (size_t)gridDim.x * blockDim.x;
  for (; i < n; i += stride) {
    int t = (int)(i / DMODEL);
    resid[i] += (acc[i] + sgate[t] * sh[i]) * invs;
  }
}

// ---------------------------------------------------------------------------
extern "C" void kernel_launch(void* const* d_in, const int* in_sizes, int n_in,
                              void* d_out, int out_size, void* d_ws, size_t ws_size,
                              hipStream_t stream)
{
  (void)in_sizes; (void)n_in; (void)out_size; (void)ws_size;
  const int*   ids = (const int*)d_in[0];
  const int*   pos = (const int*)d_in[1];
  const float* emb = (const float*)d_in[2];
  const float* qw  = (const float*)d_in[3];
  const float* kw  = (const float*)d_in[4];
  const float* vw  = (const float*)d_in[5];
  const float* ow  = (const float*)d_in[6];
  const float* ln1 = (const float*)d_in[7];
  const float* ln2 = (const float*)d_in[8];
  const float* rw  = (const float*)d_in[9];
  const float* sgw = (const float*)d_in[10];
  const float* wg  = (const float*)d_in[11];
  const float* wu  = (const float*)d_in[12];
  const float* wd  = (const float*)d_in[13];
  const float* swg = (const float*)d_in[14];
  const float* swu = (const float*)d_in[15];
  const float* swd = (const float*)d_in[16];
  const float* fln = (const float*)d_in[17];
  float* out = (float*)d_out;

  char* ws = (char*)d_ws;
  const size_t SZ_TD_F   = (size_t)NTOK * DMODEL * 4;   // 16 MB
  const size_t SZ_TD_BF  = (size_t)NTOK * DMODEL * 2;   //  8 MB
  const size_t SZ_TFS_F  = (size_t)NTOK * DFFS * 4;     // ~46 MB
  const size_t SZ_TFS_BF = (size_t)NTOK * DFFS * 2;     // ~23 MB
  const size_t SZ_WSCR   = (size_t)DMODEL * DFFS * 2;   // ~23 MB (largest transposed weight)

  size_t off = 0;
  auto take = [&](size_t b) { size_t o = off; off += (b + 255) & ~(size_t)255; return o; };
  float*          RESID  = (float*)(ws + take(SZ_TD_F));
  unsigned short* XBF    = (unsigned short*)(ws + take(SZ_TD_BF));
  float*          TMP    = (float*)(ws + take(SZ_TD_F));
  unsigned short* WSCR   = (unsigned short*)(ws + take(SZ_WSCR));
  unsigned short* ACTBF  = (unsigned short*)(ws + take(SZ_TFS_BF));
  float*          MOEACC = (float*)(ws + take(SZ_TD_F));
  float*          SH     = (float*)(ws + take(SZ_TD_F));
  float*          COMB   = (float*)(ws + take((size_t)NTOK * NEXP * 4));
  float*          SGATE  = (float*)(ws + take((size_t)NTOK * 4));
  char*           ARENA  = ws + take(2 * SZ_TFS_F);
  // attention-phase aliases inside ARENA (80 MB used of ~92 MB)
  float*          QF  = (float*)(ARENA);
  float*          KF  = (float*)(ARENA + SZ_TD_F);
  float*          VF  = (float*)(ARENA + 2 * SZ_TD_F);
  unsigned short* QBF = (unsigned short*)(ARENA + 3 * SZ_TD_F);
  unsigned short* KBF = (unsigned short*)(ARENA + 3 * SZ_TD_F + SZ_TD_BF);
  unsigned short* VT  = (unsigned short*)(ARENA + 3 * SZ_TD_F + 2 * SZ_TD_BF);
  unsigned short* PBF = (unsigned short*)(ARENA + 3 * SZ_TD_F + 3 * SZ_TD_BF);
  float*          SCORES = QF;                  // QF dead after rope
  float*          OF     = KF;                  // KF dead after rope
  unsigned short* OBF    = (unsigned short*)VF; // VF dead after V transpose
  // moe-phase aliases (attention buffers dead by then)
  float* GF = (float*)(ARENA);
  float* UF = (float*)(ARENA + SZ_TFS_F);

  auto gemm = [&](const unsigned short* A, int lda, const unsigned short* Bt, int ldb,
                  float* C, int ldc, int M, int N, int Kd) {
    int tiles = (M / 32) * (N / 32);
    gemm_bf16_wmma<<<dim3((tiles + 7) / 8), dim3(256), 0, stream>>>(A, lda, Bt, ldb, C, ldc, M, N, Kd);
  };
  auto transpose = [&](const float* src, unsigned short* dst, int rows, int cols) {
    transpose_cvt<<<dim3((cols + 31) / 32, (rows + 31) / 32), dim3(32, 8), 0, stream>>>(src, dst, rows, cols);
  };

  embed_gather<<<NTOK, 256, 0, stream>>>(ids, emb, RESID);

  for (int l = 0; l < NLAYER; ++l) {
    const float* qw_l  = qw  + (size_t)l * DMODEL * DQKV;
    const float* kw_l  = kw  + (size_t)l * DMODEL * DQKV;
    const float* vw_l  = vw  + (size_t)l * DMODEL * DQKV;
    const float* ow_l  = ow  + (size_t)l * DQKV * DMODEL;
    const float* swg_l = swg + (size_t)l * DMODEL * DFFS;
    const float* swu_l = swu + (size_t)l * DMODEL * DFFS;
    const float* swd_l = swd + (size_t)l * DFFS * DMODEL;

    // ---- attention ----
    rmsnorm<<<NTOK, 256, 0, stream>>>(RESID, ln1 + (size_t)l * DMODEL, XBF, (float*)nullptr);

    transpose(qw_l, WSCR, DMODEL, DQKV);
    gemm(XBF, DMODEL, WSCR, DMODEL, QF, DQKV, NTOK, DQKV, DMODEL);
    transpose(kw_l, WSCR, DMODEL, DQKV);
    gemm(XBF, DMODEL, WSCR, DMODEL, KF, DQKV, NTOK, DQKV, DMODEL);
    transpose(vw_l, WSCR, DMODEL, DQKV);
    gemm(XBF, DMODEL, WSCR, DMODEL, VF, DQKV, NTOK, DQKV, DMODEL);

    rope_cvt<<<NTOK * NH, 64, 0, stream>>>(QF, KF, pos, QBF, KBF);
    transpose(VF, VT, NTOK, DQKV);   // VT: (DQKV x NTOK) bf16

    for (int h = 0; h < NH; ++h) {
      // scores = Q_h (T x 128) * K_h^T  -> Bt is K_h stored (s, d): already "N x K"
      gemm(QBF + h * DHEAD, DQKV, KBF + h * DHEAD, DQKV, SCORES, NTOK, NTOK, NTOK, DHEAD);
      softmax_causal<<<NTOK, 256, 0, stream>>>(SCORES, PBF);
      // O_h = P (T x T) * V_h (T x 128); Bt = VT slice (128 x T)
      gemm(PBF, NTOK, VT + (size_t)h * DHEAD * NTOK, NTOK, OF + h * DHEAD, DQKV,
           NTOK, DHEAD, NTOK);
    }

    cvt_f32_bf16<<<2048, 256, 0, stream>>>(OF, OBF, (size_t)NTOK * DQKV);
    transpose(ow_l, WSCR, DQKV, DMODEL);
    gemm(OBF, DQKV, WSCR, DQKV, TMP, DMODEL, NTOK, DMODEL, DQKV);
    add_inplace<<<4096, 256, 0, stream>>>(RESID, TMP, (size_t)NTOK * DMODEL);

    // ---- MoE ----
    rmsnorm<<<NTOK, 256, 0, stream>>>(RESID, ln2 + (size_t)l * DMODEL, XBF, (float*)nullptr);
    router_kernel<<<NTOK, 256, 0, stream>>>(XBF, rw + (size_t)l * DMODEL * NEXP,
                                            sgw + (size_t)l * DMODEL, COMB, SGATE);

    // shared expert
    transpose(swg_l, WSCR, DMODEL, DFFS);
    gemm(XBF, DMODEL, WSCR, DMODEL, GF, DFFS, NTOK, DFFS, DMODEL);
    transpose(swu_l, WSCR, DMODEL, DFFS);
    gemm(XBF, DMODEL, WSCR, DMODEL, UF, DFFS, NTOK, DFFS, DMODEL);
    silu_mul<<<4096, 256, 0, stream>>>(GF, UF, ACTBF, (size_t)NTOK * DFFS);
    transpose(swd_l, WSCR, DFFS, DMODEL);
    gemm(ACTBF, DFFS, WSCR, DFFS, SH, DMODEL, NTOK, DMODEL, DFFS);

    // routed experts (dense, weighted by top-2 combine)
    for (int e = 0; e < NEXP; ++e) {
      const float* wg_le = wg + ((size_t)l * NEXP + e) * DMODEL * DFF;
      const float* wu_le = wu + ((size_t)l * NEXP + e) * DMODEL * DFF;
      const float* wd_le = wd + ((size_t)l * NEXP + e) * DFF * DMODEL;
      transpose(wg_le, WSCR, DMODEL, DFF);
      gemm(XBF, DMODEL, WSCR, DMODEL, GF, DFF, NTOK, DFF, DMODEL);
      transpose(wu_le, WSCR, DMODEL, DFF);
      gemm(XBF, DMODEL, WSCR, DMODEL, UF, DFF, NTOK, DFF, DMODEL);
      silu_mul<<<4096, 256, 0, stream>>>(GF, UF, ACTBF, (size_t)NTOK * DFF);
      transpose(wd_le, WSCR, DFF, DMODEL);
      gemm(ACTBF, DFF, WSCR, DFF, TMP, DMODEL, NTOK, DMODEL, DFF);
      moe_accum<<<4096, 256, 0, stream>>>(MOEACC, TMP, COMB, e, e == 0 ? 1 : 0);
    }
    moe_final<<<4096, 256, 0, stream>>>(RESID, MOEACC, SH, SGATE);
  }

  rmsnorm<<<NTOK, 256, 0, stream>>>(RESID, fln, (unsigned short*)nullptr, out);
}
